// EnhancedFusion_13400297964136
// MI455X (gfx1250) — matrix operand order repaired and checked by reading, not compiled, not run on hardware
//
#include <hip/hip_runtime.h>

#define BROWS 32768
#define DDIM  2048
#define HDIM  4096
#define LN_EPS 1e-5f

typedef __attribute__((ext_vector_type(16))) __bf16 v16bf;
typedef __attribute__((ext_vector_type(8)))  __bf16 v8bf;
typedef __attribute__((ext_vector_type(8)))  float  v8f;
typedef __attribute__((ext_vector_type(4)))  unsigned int v4u;
typedef __attribute__((ext_vector_type(8)))  int v8i;
typedef __attribute__((ext_vector_type(4)))  int v4i;

#if defined(__has_builtin)
#  if __has_builtin(__builtin_amdgcn_tensor_load_to_lds) && \
      __has_builtin(__builtin_amdgcn_s_wait_tensorcnt)
#    define HAVE_TDM 1
#  endif
#endif
#ifndef HAVE_TDM
#  define HAVE_TDM 0
#endif

static __device__ __forceinline__ unsigned short f32_to_bf16(float f) {
  union { float f; unsigned u; } v; v.f = f;
  unsigned r = v.u + 0x7FFFu + ((v.u >> 16) & 1u);   // round-to-nearest-even
  return (unsigned short)(r >> 16);
}
static __device__ __forceinline__ float bf16_to_f32(unsigned short h) {
  union { unsigned u; float f; } v; v.u = ((unsigned)h) << 16;
  return v.f;
}

// ---------------------------------------------------------------------------
// TDM 2D tile load: rows x 32 bf16 elements, row stride = kdim elements,
// into LDS with 16B padding per 64B row (LDS row stride = 80B = 40 ushorts).
// D# bitfields per CDNA5 ISA.
// This toolchain exposes the 6-arg builtin: (g0 x4u, g1 x8i, g2 x4i, g3 x4i,
// g4 x8i, cpol) — groups 2..4 unused for 2D tiles (tile_dim2=0) -> zeros.
// ---------------------------------------------------------------------------
#if HAVE_TDM
static __device__ __forceinline__ void tdm_load_tile(const unsigned short* g,
                                                     const unsigned short* lds,
                                                     int rows, int kdim) {
  const unsigned long long ga = (unsigned long long)(uintptr_t)g;
  const unsigned lo = (unsigned)(uintptr_t)lds;     // flat LDS addr: low 32 = offset
  v4u g0;
  g0.x = 0x1u;                                      // count=1, user descriptor
  g0.y = lo;                                        // lds_addr
  g0.z = (unsigned)(ga & 0xFFFFFFFFu);              // global_addr[31:0]
  g0.w = (unsigned)((ga >> 32) & 0x1FFFFFFu) | (2u << 30);  // addr[56:32] | type=2
  v8i g1;
  // mask=0 | data_size=2B (1<<16) | pad_enable (1<<20) | pad_interval=16dw (3<<22)
  // | pad_amount=4dw (3<<25)
  g1[0] = (1 << 16) | (1 << 20) | (3 << 22) | (3 << 25);
  g1[1] = (kdim & 0xFFFF) << 16;                    // tensor_dim0[15:0] (barrier addr 0)
  g1[2] = ((kdim >> 16) & 0xFFFF) | ((rows & 0xFFFF) << 16);  // dim0 hi | dim1 lo
  g1[3] = ((rows >> 16) & 0xFFFF) | (32 << 16);     // dim1 hi | tile_dim0=32
  g1[4] = rows & 0xFFFF;                            // tile_dim1=rows, tile_dim2=0
  g1[5] = kdim;                                     // tensor_dim0_stride lo
  g1[6] = 0;                                        // stride hi | dim1_stride lo
  g1[7] = 0;
  const v4i gz4 = {0, 0, 0, 0};
  const v8i gz8 = {0, 0, 0, 0, 0, 0, 0, 0};
  __builtin_amdgcn_tensor_load_to_lds(g0, g1, gz4, gz4, gz8, 0);
}
#endif

static __device__ __forceinline__ void coop_copy_tile(const unsigned short* g,
                                                      unsigned short* lds,
                                                      int rows, int kdim) {
  // fallback path: 4 chunks of 16B per 64B row, all 256 threads
  for (int i = threadIdx.x; i < rows * 4; i += 256) {
    const int row = i >> 2, q = i & 3;
    *(v8bf*)(lds + row * 40 + q * 8) = *(const v8bf*)(g + (long)row * kdim + q * 8);
  }
}

// ---------------------------------------------------------------------------
// Shared GEMM core: block tile 256(M) x 128(N), 8 waves x (32M x 128N) each,
// K step 32, TDM double-buffered LDS staging for A and B tiles.
//   A: [B][KDIM] bf16 row-major, B: [N][KDIM] bf16 (pre-transposed weights)
//   acc[nt*2+s]: WMMA 16x16 f32 tile (s = M subtile 0/1, nt = N tile 0..7)
// ---------------------------------------------------------------------------
template <int KDIM>
static __device__ __forceinline__ void gemm_core(const unsigned short* __restrict__ A,
                                                 const unsigned short* __restrict__ Bt,
                                                 v8f* acc) {
  __shared__ unsigned short ldsA[2][256 * 40];   // 40KB  (80B padded rows)
  __shared__ unsigned short ldsB[2][128 * 40];   // 20KB
  const int lane = threadIdx.x & 31;
  const int wave = threadIdx.x >> 5;
  const long am0 = (long)blockIdx.x * 256;
  const long bn0 = (long)blockIdx.y * 128;
  constexpr int KITERS = KDIM / 32;

  // fragment LDS offsets (ushort units), per CDNA5 wave32 WMMA layouts:
  //  A 16x32: lane<16 row=lane K{0..7,16..23}; lane>=16 row=lane-16 K{8..15,24..31}
  //  B 32x16: lane<16 col=lane K{0..7,16..23}; lane>=16 col=lane-16 K{8..15,24..31}
  const int aoff = (wave * 32 + (lane & 15)) * 40 + (lane >> 4) * 8;
  const int boff = (lane & 15) * 40 + (lane >> 4) * 8;

#if HAVE_TDM
  const bool issuer = (threadIdx.x < 32);          // TDM ignores EXEC; gate by wave
  if (issuer) {
    tdm_load_tile(A + am0 * KDIM, &ldsA[0][0], 256, KDIM);
    tdm_load_tile(Bt + bn0 * KDIM, &ldsB[0][0], 128, KDIM);
  }
  for (int ki = 0; ki < KITERS; ++ki) {
    const int cur = ki & 1;
    if (issuer) __builtin_amdgcn_s_wait_tensorcnt(0);  // current tiles landed
    __syncthreads();                                   // publish LDS, retire old reads
    if (issuer && (ki + 1 < KITERS)) {                 // prefetch next K-slab
      const long kb = (long)(ki + 1) * 32;
      tdm_load_tile(A + am0 * KDIM + kb, &ldsA[cur ^ 1][0], 256, KDIM);
      tdm_load_tile(Bt + bn0 * KDIM + kb, &ldsB[cur ^ 1][0], 128, KDIM);
    }
    const unsigned short* la = &ldsA[cur][0];
    const unsigned short* lb = &ldsB[cur][0];
#else
  for (int ki = 0; ki < KITERS; ++ki) {
    const long kb = (long)ki * 32;
    __syncthreads();
    coop_copy_tile(A + am0 * KDIM + kb, &ldsA[0][0], 256, KDIM);
    coop_copy_tile(Bt + bn0 * KDIM + kb, &ldsB[0][0], 128, KDIM);
    __syncthreads();
    const unsigned short* la = &ldsA[0][0];
    const unsigned short* lb = &ldsB[0][0];
#endif
    union { v16bf v; v8bf h[2]; } af[2];
#pragma unroll
    for (int s = 0; s < 2; ++s) {
      af[s].h[0] = *(const v8bf*)(la + aoff + s * 16 * 40);
      af[s].h[1] = *(const v8bf*)(la + aoff + s * 16 * 40 + 16);
    }
#pragma unroll
    for (int nt = 0; nt < 8; ++nt) {
      union { v16bf v; v8bf h[2]; } bf;
      bf.h[0] = *(const v8bf*)(lb + boff + nt * 16 * 40);
      bf.h[1] = *(const v8bf*)(lb + boff + nt * 16 * 40 + 16);
      acc[nt * 2 + 0] = __builtin_amdgcn_wmma_f32_16x16x32_bf16(
          false, af[0].v, false, bf.v, (short)0, acc[nt * 2 + 0], false, false);
      acc[nt * 2 + 1] = __builtin_amdgcn_wmma_f32_16x16x32_bf16(
          false, af[1].v, false, bf.v, (short)0, acc[nt * 2 + 1], false, false);
    }
  }
}

// ---------------------------------------------------------------------------
// 1) pack concat(x1,x2) -> bf16 combined [B][D]
// ---------------------------------------------------------------------------
__global__ void k_pack_combined(const float* __restrict__ x1,
                                const float* __restrict__ x2,
                                unsigned short* __restrict__ cb) {
  int idx = blockIdx.x * blockDim.x + threadIdx.x;
  const int total = BROWS * DDIM / 4;
  for (; idx < total; idx += gridDim.x * blockDim.x) {
    const int e = idx * 4;
    const int row = e >> 11;
    const int col = e & (DDIM - 1);
    const float* src = (col < DDIM / 2)
                         ? (x1 + (long)row * (DDIM / 2) + col)
                         : (x2 + (long)row * (DDIM / 2) + col - DDIM / 2);
    const float4 v = *(const float4*)src;
    ushort4 o;
    o.x = f32_to_bf16(v.x); o.y = f32_to_bf16(v.y);
    o.z = f32_to_bf16(v.z); o.w = f32_to_bf16(v.w);
    *(ushort4*)(cb + e) = o;
  }
}

// ---------------------------------------------------------------------------
// 2) LDS-tiled transpose + f32->bf16:  in[R][C] -> out[C][R]
// ---------------------------------------------------------------------------
__global__ void k_transpose_to_bf16(const float* __restrict__ in,
                                    unsigned short* __restrict__ out,
                                    int R, int C) {
  __shared__ float tile[32][33];
  const int bx = blockIdx.x * 32, by = blockIdx.y * 32;
  const int tx = threadIdx.x, ty = threadIdx.y;
#pragma unroll
  for (int j = 0; j < 4; ++j)
    tile[ty + j * 8][tx] = in[(long)(by + ty + j * 8) * C + bx + tx];
  __syncthreads();
#pragma unroll
  for (int j = 0; j < 4; ++j)
    out[(long)(bx + ty + j * 8) * R + by + tx] = f32_to_bf16(tile[tx][ty + j * 8]);
}

// ---------------------------------------------------------------------------
// 3) GEMM1: h = combined @ W1^T + b1 -> bf16 [B][H]
// ---------------------------------------------------------------------------
__global__ void __launch_bounds__(256)
k_gemm1(const unsigned short* __restrict__ A, const unsigned short* __restrict__ Bt,
        const float* __restrict__ bias, unsigned short* __restrict__ Hout) {
  v8f acc[16];
#pragma unroll
  for (int i = 0; i < 16; ++i)
    acc[i] = (v8f){0.f, 0.f, 0.f, 0.f, 0.f, 0.f, 0.f, 0.f};
  gemm_core<DDIM>(A, Bt, acc);

  const int lane = threadIdx.x & 31, wave = threadIdx.x >> 5;
  const int m0 = blockIdx.x * 256 + wave * 32;
  const int n0 = blockIdx.y * 128;
  const int ncol = lane & 15, rbase = (lane >> 4) * 8;
#pragma unroll
  for (int nt = 0; nt < 8; ++nt) {
    const int n = n0 + nt * 16 + ncol;
    const float bv = bias[n];
#pragma unroll
    for (int s = 0; s < 2; ++s)
#pragma unroll
      for (int r = 0; r < 8; ++r)
        Hout[(long)(m0 + s * 16 + rbase + r) * HDIM + n] =
            f32_to_bf16(acc[nt * 2 + s][r] + bv);
  }
}

// ---------------------------------------------------------------------------
// 4) LayerNorm + ReLU in place on h (bf16), one block per row
// ---------------------------------------------------------------------------
__global__ void k_ln_relu(unsigned short* __restrict__ Hb,
                          const float* __restrict__ g,
                          const float* __restrict__ b) {
  __shared__ float rs[256];
  __shared__ float rq[256];
  const int tid = threadIdx.x;
  unsigned short* hr = Hb + (long)blockIdx.x * HDIM;

  float s = 0.f, q = 0.f;
  for (int i = tid; i < HDIM; i += 256) {
    const float v = bf16_to_f32(hr[i]);
    s += v; q += v * v;
  }
  rs[tid] = s; rq[tid] = q;
  __syncthreads();
  for (int off = 128; off > 0; off >>= 1) {
    if (tid < off) { rs[tid] += rs[tid + off]; rq[tid] += rq[tid + off]; }
    __syncthreads();
  }
  const float mu  = rs[0] * (1.0f / HDIM);
  const float var = rq[0] * (1.0f / HDIM) - mu * mu;
  const float inv = rsqrtf(var + LN_EPS);
  for (int i = tid; i < HDIM; i += 256) {
    float a = (bf16_to_f32(hr[i]) - mu) * inv * g[i] + b[i];
    a = a > 0.f ? a : 0.f;
    hr[i] = f32_to_bf16(a);
  }
}

// ---------------------------------------------------------------------------
// 5) GEMM2: weights = sigmoid(act @ W2^T + b2) -> d_out, fused W3 partials
// ---------------------------------------------------------------------------
__global__ void __launch_bounds__(256)
k_gemm2(const unsigned short* __restrict__ A, const unsigned short* __restrict__ Bt,
        const float* __restrict__ bias,
        const float* __restrict__ x1, const float* __restrict__ x2,
        const float* __restrict__ W3,
        float* __restrict__ wout, float* __restrict__ part) {
  v8f acc[16];
#pragma unroll
  for (int i = 0; i < 16; ++i)
    acc[i] = (v8f){0.f, 0.f, 0.f, 0.f, 0.f, 0.f, 0.f, 0.f};
  gemm_core<HDIM>(A, Bt, acc);

  const int lane = threadIdx.x & 31, wave = threadIdx.x >> 5;
  const int m0 = blockIdx.x * 256 + wave * 32;
  const int n0 = blockIdx.y * 128;
  const int ncol = lane & 15, rbase = (lane >> 4) * 8;

  float fc[2][8][4];
#pragma unroll
  for (int s = 0; s < 2; ++s)
#pragma unroll
    for (int r = 0; r < 8; ++r)
#pragma unroll
      for (int c = 0; c < 4; ++c) fc[s][r][c] = 0.f;

#pragma unroll
  for (int nt = 0; nt < 8; ++nt) {
    const int n = n0 + nt * 16 + ncol;
    const float bv = bias[n];
    const float4 w3v = *(const float4*)(W3 + (long)n * 4);
#pragma unroll
    for (int s = 0; s < 2; ++s)
#pragma unroll
      for (int r = 0; r < 8; ++r) {
        const int mrow = m0 + s * 16 + rbase + r;
        const float z = acc[nt * 2 + s][r] + bv;
        const float w = 1.0f / (1.0f + __expf(-z));      // sigmoid
        wout[(long)mrow * DDIM + n] = w;
        const float cv = (n < DDIM / 2)
                           ? x1[(long)mrow * (DDIM / 2) + n]
                           : x2[(long)mrow * (DDIM / 2) + n - DDIM / 2];
        const float wc = w * cv;
        fc[s][r][0] += wc * w3v.x; fc[s][r][1] += wc * w3v.y;
        fc[s][r][2] += wc * w3v.z; fc[s][r][3] += wc * w3v.w;
      }
  }
  // reduce across 16 lanes sharing the same rows (wave32 width-16 groups)
#pragma unroll
  for (int mask = 8; mask >= 1; mask >>= 1)
#pragma unroll
    for (int s = 0; s < 2; ++s)
#pragma unroll
      for (int r = 0; r < 8; ++r)
#pragma unroll
        for (int c = 0; c < 4; ++c)
          fc[s][r][c] += __shfl_xor(fc[s][r][c], mask, 16);

  if ((lane & 15) == 0) {  // lane0 -> rows +0..7, lane16 -> rows +8..15
    float* pb = part + (long)blockIdx.y * BROWS * 4;
#pragma unroll
    for (int s = 0; s < 2; ++s)
#pragma unroll
      for (int r = 0; r < 8; ++r)
#pragma unroll
        for (int c = 0; c < 4; ++c)
          pb[(long)(m0 + s * 16 + rbase + r) * 4 + c] = fc[s][r][c];
  }
}

// ---------------------------------------------------------------------------
// 6) final_class = b3 + sum over 16 column-block partials (deterministic)
// ---------------------------------------------------------------------------
__global__ void k_final(const float* __restrict__ part,
                        const float* __restrict__ b3,
                        float* __restrict__ out) {
  const int idx = blockIdx.x * blockDim.x + threadIdx.x;
  if (idx >= BROWS * 4) return;
  float s = b3[idx & 3];
#pragma unroll
  for (int bn = 0; bn < 16; ++bn) s += part[(long)bn * BROWS * 4 + idx];
  out[idx] = s;
}

// ---------------------------------------------------------------------------
extern "C" void kernel_launch(void* const* d_in, const int* in_sizes, int n_in,
                              void* d_out, int out_size, void* d_ws, size_t ws_size,
                              hipStream_t stream) {
  const float* x1   = (const float*)d_in[0];
  const float* x2   = (const float*)d_in[1];
  const float* W1   = (const float*)d_in[2];
  const float* b1   = (const float*)d_in[3];
  const float* ln_g = (const float*)d_in[4];
  const float* ln_b = (const float*)d_in[5];
  const float* W2   = (const float*)d_in[6];
  const float* b2   = (const float*)d_in[7];
  const float* W3   = (const float*)d_in[8];
  const float* b3   = (const float*)d_in[9];

  float* fc_out = (float*)d_out;                    // final_class [B,4]
  float* w_out  = fc_out + (size_t)BROWS * 4;       // weights [B,D]

  char* ws = (char*)d_ws;
  size_t off = 0;
  auto suballoc = [&](size_t bytes) -> void* {
    void* p = ws + off;
    off = (off + bytes + 255) & ~(size_t)255;
    return p;
  };
  unsigned short* cb  = (unsigned short*)suballoc((size_t)BROWS * DDIM * 2);
  unsigned short* w1t = (unsigned short*)suballoc((size_t)HDIM * DDIM * 2);
  unsigned short* w2t = (unsigned short*)suballoc((size_t)DDIM * HDIM * 2);
  unsigned short* hb  = (unsigned short*)suballoc((size_t)BROWS * HDIM * 2);
  float*          pb  = (float*)suballoc((size_t)16 * BROWS * 4 * 4);

  hipLaunchKernelGGL(k_pack_combined, dim3(4096), dim3(256), 0, stream, x1, x2, cb);
  hipLaunchKernelGGL(k_transpose_to_bf16, dim3(HDIM / 32, DDIM / 32), dim3(32, 8),
                     0, stream, W1, w1t, DDIM, HDIM);
  hipLaunchKernelGGL(k_transpose_to_bf16, dim3(DDIM / 32, HDIM / 32), dim3(32, 8),
                     0, stream, W2, w2t, HDIM, DDIM);
  hipLaunchKernelGGL(k_gemm1, dim3(BROWS / 256, HDIM / 128), dim3(256),
                     0, stream, cb, w1t, b1, hb);
  hipLaunchKernelGGL(k_ln_relu, dim3(BROWS), dim3(256), 0, stream, hb, ln_g, ln_b);
  hipLaunchKernelGGL(k_gemm2, dim3(BROWS / 256, DDIM / 128), dim3(256),
                     0, stream, hb, w2t, b2, x1, x2, W3, w_out, pb);
  hipLaunchKernelGGL(k_final, dim3((BROWS * 4 + 255) / 256), dim3(256),
                     0, stream, pb, b3, fc_out);
}
